// PershomBase_31705448579352
// MI455X (gfx1250) — matrix operand order repaired and checked by reading, not compiled, not run on hardware
//
#include <hip/hip_runtime.h>
#include <hip/hip_bf16.h>

// Problem constants (from reference): G=64, C=32, L=128, D=64, H=128, NCLS=10
#define GG   64
#define CC   32
#define LL   128
#define DD   64
#define HH   128
#define NCLS 10
#define BB   (GG * CC)          // 2048 sequences
#define KTOT (DD + HH)          // 192 combined K (input | hidden)
#define NG   (4 * HH)           // 512 gate columns

typedef __attribute__((ext_vector_type(16))) __bf16 v16bf;
typedef __attribute__((ext_vector_type(8)))  float  v8f;

union Frag16 { v16bf v; unsigned int u[8]; };

__device__ __forceinline__ unsigned short f2bf(float f) {
    union { float f; unsigned int u; } x; x.f = f;
    unsigned int u = x.u;
    u += 0x7FFFu + ((u >> 16) & 1u);     // round-to-nearest-even
    return (unsigned short)(u >> 16);
}

__device__ __forceinline__ float fast_sigmoid(float x) {
    return 1.0f / (1.0f + __expf(-x));
}

__device__ __forceinline__ float fast_tanh(float x) {
    float ax = fabsf(x);
    float e  = __expf(-2.0f * ax);       // in (0,1], no overflow
    float t  = (1.0f - e) / (1.0f + e);
    return copysignf(t, x);
}

// ---------------------------------------------------------------------------
// Fused LSTM kernel: one block owns a 16-row batch tile, runs the full L=128
// sequential loop with register-resident weights (bf16 B-fragments) and
// LDS-resident x_t/h pane. Emits v_wmma_f32_16x16x32_bf16 for all GEMM work.
// Output: rep[b][h] = sum_t h_t  (f32, [2048 x 128]) into workspace.
// ---------------------------------------------------------------------------
__global__ __launch_bounds__(512) void lstm_fused_wmma_kernel(
    const float* __restrict__ x,      // [B, L, D]
    const float* __restrict__ W_ih,   // [4H, D]
    const float* __restrict__ W_hh,   // [4H, H]
    const float* __restrict__ b_ih,   // [4H]
    const float* __restrict__ b_hh,   // [4H]
    float* __restrict__ rep)          // [B, H]
{
    __shared__ unsigned short xh[16][KTOT];   // bf16 [x_t | h], 6 KB
    __shared__ float          gates[16][NG];  // f32 gate pre-acts, 32 KB
    __shared__ float          bias_s[NG];     // 2 KB

    const int tid  = threadIdx.x;             // 0..511 (16 waves)
    const int lane = tid & 31;
    const int wv   = tid >> 5;                // wave id 0..15
    const int b0   = blockIdx.x * 16;         // batch tile base

    // WMMA lane geometry
    const int lrow = lane & 15;               // A row / C column-in-tile
    const int lg   = lane >> 4;               // lane group 0/1
    const int n0   = wv * 32;                 // this wave's first gate column

    // Elementwise mapping: thread owns 4 consecutive h-columns of one row
    const int erow = tid >> 5;                // 0..15
    const int ecol = (tid * 4) & (HH - 1);    // 0..124

    // ---- one-time setup: bias to LDS, zero h-pane, pack B fragments -------
    if (tid < NG) bias_s[tid] = b_ih[tid] + b_hh[tid];
    {   // zero h region of xh (16 x 128 bf16)
        unsigned short* hz = &xh[erow][DD + ecol];
        hz[0] = 0; hz[1] = 0; hz[2] = 0; hz[3] = 0;
    }

    // Register-resident combined weight [W_ih | W_hh]^T as WMMA B fragments:
    // 2 N-tiles x 6 K-frags x 8 VGPRs = 96 VGPRs per wave, loaded once.
    Frag16 bfr[2][6];
#pragma unroll
    for (int nt = 0; nt < 2; ++nt) {
        const int col = n0 + nt * 16 + lrow;  // gate row index (0..511)
#pragma unroll
        for (int kf = 0; kf < 6; ++kf) {
#pragma unroll
            for (int j = 0; j < 8; ++j) {
                const int k = kf * 32 + lg * 16 + j * 2;   // B k order used here: VGPRj K=2j(+g*16)
                float f0, f1;
                if (k < DD) { f0 = W_ih[col * DD + k];        f1 = W_ih[col * DD + k + 1]; }
                else        { f0 = W_hh[col * HH + (k - DD)]; f1 = W_hh[col * HH + (k - DD) + 1]; }
                bfr[nt][kf].u[j] = (unsigned int)f2bf(f0) | ((unsigned int)f2bf(f1) << 16);
            }
        }
    }

    float c_reg[4] = {0.f, 0.f, 0.f, 0.f};
    float h_sum[4] = {0.f, 0.f, 0.f, 0.f};

    // x staging mapping: 512 threads x 2 floats = 16 rows x 64 cols
    const int sxrow = tid >> 5;               // tid*2/64
    const int sxcol = (tid * 2) & (DD - 1);
    const float* xbase = x + ((size_t)(b0 + sxrow) * LL) * DD + sxcol;

    for (int t = 0; t < LL; ++t) {
        // ---- stage x_t into LDS pane as bf16 ----
        {
            const float* xp = xbase + (size_t)t * DD;
            xh[sxrow][sxcol]     = f2bf(xp[0]);
            xh[sxrow][sxcol + 1] = f2bf(xp[1]);
            if (t + 1 < LL)
                __builtin_prefetch(xp + DD, 0, 3);   // global_prefetch_b8 for next step
        }
        __syncthreads();   // x_t and previous h visible

        // ---- WMMA phase: gates[16 x 32cols] += A(16x192) x B(192x32) ----
        v8f acc0 = {}; v8f acc1 = {};
#pragma unroll
        for (int kf = 0; kf < 6; ++kf) {
            Frag16 a;
#pragma unroll
            for (int j = 0; j < 8; ++j) {
                // A k order used here (contiguous): VGPRj -> K = g*16 + 2j (matches B fragments)
                const int kl = lg * 16 + j * 2;
                a.u[j] = *reinterpret_cast<const unsigned int*>(&xh[lrow][kf * 32 + kl]);
            }
            acc0 = __builtin_amdgcn_wmma_f32_16x16x32_bf16(
                       false, a.v, false, bfr[0][kf].v, (short)0, acc0, false, false);
            acc1 = __builtin_amdgcn_wmma_f32_16x16x32_bf16(
                       false, a.v, false, bfr[1][kf].v, (short)0, acc1, false, false);
        }
        // C/D layout: VGPR r -> M = r + 8*lanegroup, column = n0 + (lane&15)
#pragma unroll
        for (int r = 0; r < 8; ++r) {
            const int m = r + 8 * lg;
            gates[m][n0 + lrow]      = acc0[r];
            gates[m][n0 + 16 + lrow] = acc1[r];
        }
        __syncthreads();   // gates visible; all xh reads done

        // ---- elementwise gate math; h back to LDS as bf16 ----
#pragma unroll
        for (int e = 0; e < 4; ++e) {
            const int col = ecol + e;
            const float ig = fast_sigmoid(gates[erow][col]            + bias_s[col]);
            const float fg = fast_sigmoid(gates[erow][HH + col]       + bias_s[HH + col]);
            const float gg = fast_tanh  (gates[erow][2 * HH + col]    + bias_s[2 * HH + col]);
            const float og = fast_sigmoid(gates[erow][3 * HH + col]   + bias_s[3 * HH + col]);
            c_reg[e] = fg * c_reg[e] + ig * gg;
            const float h = og * fast_tanh(c_reg[e]);
            h_sum[e] += h;
            xh[erow][DD + col] = f2bf(h);
        }
        __syncthreads();   // protect gates/xh before next iteration's writes
    }

    // ---- write per-sequence hidden-state sums ----
#pragma unroll
    for (int e = 0; e < 4; ++e)
        rep[(size_t)(b0 + erow) * HH + ecol + e] = h_sum[e];
}

// ---------------------------------------------------------------------------
// Head kernel: mean over the 32 cycles, classifier, log_softmax.  Tiny.
// One block per graph, 128 threads (one per hidden dim).
// ---------------------------------------------------------------------------
__global__ __launch_bounds__(128) void lstm_head_kernel(
    const float* __restrict__ rep,    // [B, H]
    const float* __restrict__ W_cls,  // [NCLS, H]
    const float* __restrict__ b_cls,  // [NCLS]
    float* __restrict__ out)          // [G, NCLS]
{
    __shared__ float mean_s[HH];
    __shared__ float logit_s[NCLS];

    const int g = blockIdx.x;
    const int h = threadIdx.x;

    float s = 0.f;
#pragma unroll 4
    for (int c = 0; c < CC; ++c)
        s += rep[((size_t)(g * CC + c)) * HH + h];
    mean_s[h] = s * (1.0f / (float)CC);
    __syncthreads();

    if (h < NCLS) {
        float acc = b_cls[h];
#pragma unroll 4
        for (int k = 0; k < HH; ++k)
            acc += mean_s[k] * W_cls[h * HH + k];
        logit_s[h] = acc;
    }
    __syncthreads();

    if (h == 0) {
        float m = logit_s[0];
        for (int n = 1; n < NCLS; ++n) m = fmaxf(m, logit_s[n]);
        float se = 0.f;
        for (int n = 0; n < NCLS; ++n) se += __expf(logit_s[n] - m);
        const float lse = m + __logf(se);
        for (int n = 0; n < NCLS; ++n) out[g * NCLS + n] = logit_s[n] - lse;
    }
}

extern "C" void kernel_launch(void* const* d_in, const int* in_sizes, int n_in,
                              void* d_out, int out_size, void* d_ws, size_t ws_size,
                              hipStream_t stream) {
    const float* x     = (const float*)d_in[0];  // cycle_reps [G,C,L,D]
    const float* W_ih  = (const float*)d_in[1];  // [4H, D]
    const float* W_hh  = (const float*)d_in[2];  // [4H, H]
    const float* b_ih  = (const float*)d_in[3];  // [4H]
    const float* b_hh  = (const float*)d_in[4];  // [4H]
    const float* W_cls = (const float*)d_in[5];  // [NCLS, H]
    const float* b_cls = (const float*)d_in[6];  // [NCLS]

    float* rep = (float*)d_ws;                   // [B, H] f32 = 1 MB scratch

    lstm_fused_wmma_kernel<<<BB / 16, 512, 0, stream>>>(x, W_ih, W_hh, b_ih, b_hh, rep);
    lstm_head_kernel<<<GG, HH, 0, stream>>>(rep, W_cls, b_cls, (float*)d_out);
}
